// POS_31525059953155
// MI455X (gfx1250) — matrix operand order repaired and checked by reading, not compiled, not run on hardware
//
#include <hip/hip_runtime.h>
#include <math.h>

typedef __attribute__((ext_vector_type(16))) _Float16 v16h;
typedef __attribute__((ext_vector_type(8)))  _Float16 v8h;
typedef __attribute__((ext_vector_type(8)))  float    v8f;

#define B_     32768
#define W_     5
#define L_     20
#define CE_    32
#define OC_    30
#define KW_    3
#define WE_    50
#define HID_   100
#define TAGS_  36
#define FEAT_  400
#define FEATP_ 416   // 13*32, K-padded feature width (f16)
#define HIDP_  128   // 4*32,  K-padded hidden width (f16)
#define NWS_   (B_*W_)
#define TPOS_  18    // L - K + 1 conv output positions

// workspace layout (bytes)
#define OFF_FEATS  0
#define OFF_H      (B_*FEATP_*2)              // 27,262,976
#define OFF_CFRAG  (OFF_H + B_*HIDP_*2)       // + 8,388,608
#define OFF_F1FRAG (OFF_CFRAG + 6*512*2)
#define OFF_OFRAG  (OFF_F1FRAG + 91*512*2)

// ---- fragment load helpers -------------------------------------------------
// B-fragment / pre-packed: 32 contiguous bytes per lane
__device__ __forceinline__ v16h ld_frag32B(const _Float16* p) {
  union { v16h v; v8h h[2]; } u;
  u.h[0] = *(const v8h*)(p);
  u.h[1] = *(const v8h*)(p + 8);
  return u.v;
}
// A-fragment per ISA 16-bit A layout: two 16B chunks at p and p+16 halves
__device__ __forceinline__ v16h ld_frag_a(const _Float16* p) {
  union { v16h v; v8h h[2]; } u;
  u.h[0] = *(const v8h*)(p);
  u.h[1] = *(const v8h*)(p + 16);
  return u.v;
}
#define WMMA_F16(a,b,c) \
  __builtin_amdgcn_wmma_f32_16x16x32_f16(false,(a),false,(b),(short)0,(c),false,false)

// ---- K0: pack weights into WMMA B-fragment layout --------------------------
// B-matrix (32x16, 16-bit): lane ln holds column N = ln&15; element e -> K = (e&7) + 8*(ln>>4) + 16*(e>>3)
__global__ void pos_pack(const float* __restrict__ conv_w,
                         const float* __restrict__ fc1_w,
                         const float* __restrict__ out_w,
                         _Float16* cfrag, _Float16* f1frag, _Float16* ofrag) {
  const int tid = threadIdx.x, nthr = blockDim.x;
  // conv: contraction c = k*32+ic (so im2col rows are contiguous windows of x[l][ic])
  for (int i = tid; i < 6*512; i += nthr) {
    int f = i >> 9, rem = i & 511, ln = rem >> 4, e = rem & 15;
    int kk = f >> 1, ntc = f & 1;
    int Kc = kk*32 + (e & 7) + 8*(ln >> 4) + 16*(e >> 3);
    int oc = ntc*16 + (ln & 15);
    float v = 0.f;
    if (oc < OC_) { int ic = Kc & 31, k = Kc >> 5; v = conv_w[oc*(CE_*KW_) + ic*KW_ + k]; }
    cfrag[i] = (_Float16)v;
  }
  // fc1: 13 K-steps x 7 N-tiles
  for (int i = tid; i < 91*512; i += nthr) {
    int f = i >> 9, rem = i & 511, ln = rem >> 4, e = rem & 15;
    int ks = f / 7, nn = f % 7;
    int Kc = ks*32 + (e & 7) + 8*(ln >> 4) + 16*(e >> 3);
    int n  = nn*16 + (ln & 15);
    float v = 0.f;
    if (Kc < FEAT_ && n < HID_) v = fc1_w[n*FEAT_ + Kc];
    f1frag[i] = (_Float16)v;
  }
  // out: 4 K-steps x 3 N-tiles
  for (int i = tid; i < 12*512; i += nthr) {
    int f = i >> 9, rem = i & 511, ln = rem >> 4, e = rem & 15;
    int ks = f / 3, nn = f % 3;
    int Kc = ks*32 + (e & 7) + 8*(ln >> 4) + 16*(e >> 3);
    int n  = nn*16 + (ln & 15);
    float v = 0.f;
    if (Kc < HID_ && n < TAGS_) v = out_w[n*HID_ + Kc];
    ofrag[i] = (_Float16)v;
  }
}

// ---- K1: char-CNN via WMMA + max-pool, word-embedding gather ---------------
// one wave = 16 word-samples; 2 waves / block (LDS 40 KB)
__global__ __launch_bounds__(64)
void pos_conv_feats(const int* __restrict__ input,
                    const float* __restrict__ emb,
                    const float* __restrict__ cemb,
                    const float* __restrict__ conv_b,
                    const _Float16* __restrict__ cfrag,
                    _Float16* __restrict__ feats) {
  __shared__ _Float16 xs[2][16][640];          // [wave][sample][l*32+ic], f16
  const int lane = threadIdx.x & 31;
  const int wv   = threadIdx.x >> 5;
  const int g0   = (blockIdx.x*2 + wv) * 16;   // first word-sample of this wave
  _Float16* xw = &xs[wv][0][0];

  // gather char embeddings -> LDS (char table is 12.8 KB, L2/L0 resident)
  for (int p = lane; p < 16*L_; p += 32) {
    int s = p / L_, l = p % L_;
    int g = g0 + s, b = g / W_, w = g % W_;
    int cid = input[(b*W_ + w)*(1+L_) + 1 + l];
    const float* row = cemb + cid*CE_;
    _Float16* dst = xw + s*640 + l*CE_;
#pragma unroll
    for (int ic = 0; ic < CE_; ++ic) dst[ic] = (_Float16)row[ic];
  }
  // gather word embeddings straight into feats (f16)
  for (int p = lane; p < 16*WE_; p += 32) {
    int s = p / WE_, c = p % WE_;
    int g = g0 + s, b = g / W_, w = g % W_;
    int wid = input[(b*W_ + w)*(1+L_)];
    feats[b*FEATP_ + w*80 + c] = (_Float16)emb[wid*WE_ + c];
  }
  // zero the K-padding tail of each feature row (once, when w==0 sample present)
  for (int p = lane; p < 16*16; p += 32) {
    int s = p >> 4, c = p & 15;
    int g = g0 + s;
    if (g % W_ == 0) feats[(g / W_)*FEATP_ + FEAT_ + c] = (_Float16)0.f;
  }
  __syncthreads();

  v16h bfr[6];
#pragma unroll
  for (int f = 0; f < 6; ++f) bfr[f] = ld_frag32B(cfrag + (f*32 + lane)*16);

  v8f pool0, pool1;
#pragma unroll
  for (int i = 0; i < 8; ++i) { pool0[i] = -3.0e38f; pool1[i] = -3.0e38f; }

  const int M = lane & 15, off = (lane >> 4) * 8;
  const _Float16* arow = xw + M*640 + off;
  for (int t = 0; t < TPOS_; ++t) {           // 18 positions x (3 K-steps x 2 N-tiles) WMMA
    v8f c0 = {}, c1 = {};
#pragma unroll
    for (int kk = 0; kk < 3; ++kk) {
      v16h a = ld_frag_a(arow + t*32 + kk*32); // im2col row = contiguous LDS window
      c0 = WMMA_F16(a, bfr[kk*2+0], c0);
      c1 = WMMA_F16(a, bfr[kk*2+1], c1);
    }
#pragma unroll
    for (int i = 0; i < 8; ++i) { pool0[i] = fmaxf(pool0[i], c0[i]); pool1[i] = fmaxf(pool1[i], c1[i]); }
  }
  // bias + scatter pooled (C/D layout: VGPR v, lane ln -> M=v+8*(ln>>4), N=ln&15)
#pragma unroll
  for (int nt = 0; nt < 2; ++nt) {
    int oc = nt*16 + (lane & 15);
    if (oc < OC_) {
      float bias = conv_b[oc];
      v8f pl = nt ? pool1 : pool0;
#pragma unroll
      for (int v = 0; v < 8; ++v) {
        int s = v + 8*(lane >> 4);
        int g = g0 + s, b = g / W_, w = g % W_;
        feats[b*FEATP_ + w*80 + WE_ + oc] = (_Float16)(pl[v] + bias);
      }
    }
  }
}

// ---- K2: h = tanh(feats @ fc1_w^T + b), WMMA GEMM 16x112 tile per wave -----
__global__ __launch_bounds__(128)
void pos_fc1(const _Float16* __restrict__ feats,
             const _Float16* __restrict__ f1frag,
             const float* __restrict__ fc1_b,
             _Float16* __restrict__ hbuf) {
  const int lane = threadIdx.x & 31;
  const int wv   = threadIdx.x >> 5;
  const int r0   = (blockIdx.x*4 + wv) * 16;
  const int M = lane & 15, off = (lane >> 4) * 8;
  const _Float16* arow = feats + (r0 + M)*FEATP_ + off;

  v8f acc[7] = {};
  for (int ks = 0; ks < 13; ++ks) {
    v16h a = ld_frag_a(arow + ks*32);
#pragma unroll
    for (int nt = 0; nt < 7; ++nt) {
      v16h b = ld_frag32B(f1frag + ((ks*7 + nt)*32 + lane)*16);
      acc[nt] = WMMA_F16(a, b, acc[nt]);
    }
  }
#pragma unroll
  for (int nt = 0; nt < 7; ++nt) {
    int col = nt*16 + (lane & 15);
    float bias = (col < HID_) ? fc1_b[col] : 0.f;
#pragma unroll
    for (int v = 0; v < 8; ++v) {
      int r = r0 + v + 8*(lane >> 4);
      float hv = (col < HID_) ? tanhf(acc[nt][v] + bias) : 0.f;
      hbuf[r*HIDP_ + col] = (_Float16)hv;
    }
  }
  // zero padded K columns 112..127 for the next GEMM
  for (int p = lane; p < 256; p += 32) {
    int v = p >> 4, c = 112 + (p & 15);
    hbuf[(r0 + v)*HIDP_ + c] = (_Float16)0.f;
  }
}

// ---- K3: out = h @ out_w^T + b, WMMA GEMM 16x48 tile per wave --------------
__global__ __launch_bounds__(128)
void pos_out(const _Float16* __restrict__ hbuf,
             const _Float16* __restrict__ ofrag,
             const float* __restrict__ out_b,
             float* __restrict__ out) {
  const int lane = threadIdx.x & 31;
  const int wv   = threadIdx.x >> 5;
  const int r0   = (blockIdx.x*4 + wv) * 16;
  const int M = lane & 15, off = (lane >> 4) * 8;
  const _Float16* arow = hbuf + (r0 + M)*HIDP_ + off;

  v8f acc[3] = {};
#pragma unroll
  for (int ks = 0; ks < 4; ++ks) {
    v16h a = ld_frag_a(arow + ks*32);
#pragma unroll
    for (int nt = 0; nt < 3; ++nt) {
      v16h b = ld_frag32B(ofrag + ((ks*3 + nt)*32 + lane)*16);
      acc[nt] = WMMA_F16(a, b, acc[nt]);
    }
  }
#pragma unroll
  for (int nt = 0; nt < 3; ++nt) {
    int col = nt*16 + (lane & 15);
    if (col < TAGS_) {
      float bias = out_b[col];
#pragma unroll
      for (int v = 0; v < 8; ++v) {
        int r = r0 + v + 8*(lane >> 4);
        out[r*TAGS_ + col] = acc[nt][v] + bias;
      }
    }
  }
}

extern "C" void kernel_launch(void* const* d_in, const int* in_sizes, int n_in,
                              void* d_out, int out_size, void* d_ws, size_t ws_size,
                              hipStream_t stream) {
  const int*   input = (const int*)  d_in[0];
  const float* emb   = (const float*)d_in[1];
  const float* cemb  = (const float*)d_in[2];
  const float* convw = (const float*)d_in[3];
  const float* convb = (const float*)d_in[4];
  const float* fc1w  = (const float*)d_in[5];
  const float* fc1b  = (const float*)d_in[6];
  const float* outw  = (const float*)d_in[7];
  const float* outb  = (const float*)d_in[8];
  float* out = (float*)d_out;

  char* ws = (char*)d_ws;
  _Float16* feats  = (_Float16*)(ws + OFF_FEATS);
  _Float16* hbuf   = (_Float16*)(ws + OFF_H);
  _Float16* cfrag  = (_Float16*)(ws + OFF_CFRAG);
  _Float16* f1frag = (_Float16*)(ws + OFF_F1FRAG);
  _Float16* ofrag  = (_Float16*)(ws + OFF_OFRAG);

  pos_pack<<<1, 256, 0, stream>>>(convw, fc1w, outw, cfrag, f1frag, ofrag);
  pos_conv_feats<<<NWS_/32, 64, 0, stream>>>(input, emb, cemb, convb, cfrag, feats);
  pos_fc1<<<B_/64, 128, 0, stream>>>(feats, f1frag, fc1b, hbuf);
  pos_out<<<B_/64, 128, 0, stream>>>(hbuf, ofrag, outb, out);
}
